// GIN_edge_12919261627156
// MI455X (gfx1250) — compile-verified, matrix operand order checked by reading, not run on hardware
//
#include <hip/hip_runtime.h>
#include <hip/hip_bf16.h>

// ---------------- problem constants (match reference) ----------------
#define N_NODES_C  100000
#define N_EDGES_C  1600000
#define N_GRAPHS_C 1000
#define HID_C      64
#define IN_DIM_C   7
#define EDGE_DIM_C 4
#define NUM_TASKS_C 12

typedef __attribute__((ext_vector_type(16))) _Float16 v16h;
typedef __attribute__((ext_vector_type(8)))  _Float16 v8h;
typedef __attribute__((ext_vector_type(8)))  float    v8f;

// Hardware FP32 atomic add (global_atomic_add_f32), not a CAS loop.
__device__ __forceinline__ void atom_add_f32(float* p, float v) {
    unsafeAtomicAdd(p, v);
}

// ---------------------------------------------------------------------
// WMMA operand fill from LDS.
// 16-bit A/B operand layout for V_WMMA_*_16X16X32 (CDNA5 ISA 7.12.2):
//   lane<16  -> row r=lane,    holds K = {kb+0..7,  kb+16..23}
//   lane>=16 -> row r=lane-16, holds K = {kb+8..15, kb+24..31}
// Both A (activations, row-major [16][ld]) and B (weights as Wt[n][k],
// row-major [16-col-tile][ld]) are staged in LDS so each operand fill is
// two ds_load_b128 per lane.
// ---------------------------------------------------------------------
__device__ __forceinline__ v16h lds_load_tile(const _Float16* s, int ld, int kbase) {
    const int lane = threadIdx.x & 31;
    const int r    = lane & 15;
    const int koff = (lane < 16) ? 0 : 8;
    const _Float16* p = s + r * ld + kbase + koff;
    v8h lo = *(const v8h*)(p);        // K = kb+koff+0..7
    v8h hi = *(const v8h*)(p + 16);   // K = kb+koff+16..23
    v16h out;
#pragma unroll
    for (int e = 0; e < 8; ++e) { out[e] = lo[e]; out[8 + e] = hi[e]; }
    return out;
}

// ---------------------------------------------------------------------
// Fused 2-layer node MLP:  out = relu( relu((in0+in1) @ W1^T + b1) @ W2^T + b2 )
// (the trailing relu is the post-conv relu in the reference)
// One block = 128 threads = 4 waves; block handles 16 node rows,
// each wave owns one 16-wide slice of the 64 output columns.
// TI = element type of in0 (f32 for conv1's x, f16 for conv2's h),
// TO = element type of out (f16 for h, f32 for h2).
// ---------------------------------------------------------------------
template <int K1, typename TI, typename TO>
__global__ __launch_bounds__(128)
void node_mlp_kernel(const TI* __restrict__ in0, const float* __restrict__ in1,
                     const float* __restrict__ W1,  const float* __restrict__ b1,
                     const float* __restrict__ W2,  const float* __restrict__ b2,
                     TO* __restrict__ out, int n_nodes) {
    constexpr int K1P = (K1 + 31) & ~31;          // K padded to multiple of 32

    __shared__ __align__(16) _Float16 sA  [16 * K1P];      // staged input rows
    __shared__ __align__(16) _Float16 sMid[16 * HID_C];    // layer-1 activations
    __shared__ __align__(16) _Float16 sB1 [HID_C * K1P];   // W1 as Wt[n][k] (f16)
    __shared__ __align__(16) _Float16 sB2 [HID_C * HID_C]; // W2 as Wt[n][k] (f16)

    const int row0 = blockIdx.x * 16;
    const int tid  = threadIdx.x;
    const int wave = tid >> 5;
    const int lane = tid & 31;
    const int n0   = wave * 16;                   // this wave's output-col tile

    // ---- stage A = in0 + in1 (zero-padded K) ----
    for (int i = tid; i < 16 * K1P; i += 128) {
        const int m = i / K1P, k = i - m * K1P;
        const int r = row0 + m;
        float v = 0.0f;
        if (k < K1 && r < n_nodes)
            v = (float)in0[(size_t)r * K1 + k] + in1[(size_t)r * K1 + k];
        sA[i] = (_Float16)v;
    }
    // ---- stage weights (row-major [out=64][in=K]) into f16 LDS ----
    for (int i = tid; i < HID_C * K1P; i += 128) {
        const int n = i / K1P, k = i - n * K1P;
        sB1[i] = (_Float16)((k < K1) ? W1[n * K1 + k] : 0.0f);
    }
    for (int i = tid; i < HID_C * HID_C; i += 128) {
        const int n = i >> 6, k = i & 63;
        sB2[i] = (_Float16)W2[n * HID_C + k];
    }
    __syncthreads();

    // ---- layer 1: D = A(16xK1P) x B(K1Px16) ----
    v8f acc = {};
#pragma unroll
    for (int kb = 0; kb < K1P; kb += 32) {
        v16h a = lds_load_tile(sA, K1P, kb);
        v16h b = lds_load_tile(sB1 + n0 * K1P, K1P, kb);
        acc = __builtin_amdgcn_wmma_f32_16x16x32_f16(false, a, false, b,
                                                     (short)0, acc, false, false);
    }
    {   // C/D layout: VGPR r -> M = r + (lane<16 ? 0 : 8), N = lane&15
        const int n  = lane & 15;
        const int mb = (lane < 16) ? 0 : 8;
        const float bias = b1[n0 + n];
#pragma unroll
        for (int r = 0; r < 8; ++r) {
            float v = acc[r] + bias;
            v = v > 0.0f ? v : 0.0f;
            sMid[(mb + r) * HID_C + (n0 + n)] = (_Float16)v;
        }
    }
    __syncthreads();

    // ---- layer 2: D = Mid(16x64) x B2(64x16), two k-steps ----
    v8f acc2 = {};
#pragma unroll
    for (int kb = 0; kb < HID_C; kb += 32) {
        v16h a = lds_load_tile(sMid, HID_C, kb);
        v16h b = lds_load_tile(sB2 + n0 * HID_C, HID_C, kb);
        acc2 = __builtin_amdgcn_wmma_f32_16x16x32_f16(false, a, false, b,
                                                      (short)0, acc2, false, false);
    }
    {
        const int n  = lane & 15;
        const int mb = (lane < 16) ? 0 : 8;
        const float bias = b2[n0 + n];
#pragma unroll
        for (int r = 0; r < 8; ++r) {
            const int rr = row0 + mb + r;
            if (rr < n_nodes) {
                float v = acc2[r] + bias;
                v = v > 0.0f ? v : 0.0f;            // post-conv relu
                out[(size_t)rr * HID_C + (n0 + n)] = (TO)v;
            }
        }
    }
}

// ---------------------------------------------------------------------
// conv1 message + aggregate: aggr1[dst] += relu(x[src] + edge_attr@e1W^T + e1b)
// 8 threads per edge (c = 0..7, c<7 active); coalesced atomics.
// ---------------------------------------------------------------------
__global__ __launch_bounds__(256)
void edge1_kernel(const float* __restrict__ x, const int* __restrict__ ei,
                  const float* __restrict__ ea,
                  const float* __restrict__ eW, const float* __restrict__ eb,
                  float* __restrict__ aggr) {
    const long long t = (long long)blockIdx.x * 256 + threadIdx.x;
    const long long e = t >> 3;
    const int c = (int)(t & 7);
    if (e >= N_EDGES_C || c >= IN_DIM_C) return;
    const int src = ei[e];
    const int dst = ei[N_EDGES_C + e];
    const float4 a = *(const float4*)(ea + e * EDGE_DIM_C);
    const float4 w = *(const float4*)(eW + c * EDGE_DIM_C);
    float v = eb[c] + a.x * w.x + a.y * w.y + a.z * w.z + a.w * w.w;
    v += x[(size_t)src * IN_DIM_C + c];
    v = fmaxf(v, 0.0f);
    atom_add_f32(&aggr[(size_t)dst * IN_DIM_C + c], v);
}

// ---------------------------------------------------------------------
// conv2 message + aggregate: aggr2[dst] += relu(h[src] + edge_attr@e2W^T + e2b)
// 64 threads per edge; h is f16 -> 128B coalesced gather per edge,
// f32 hardware atomics for the 256B scatter.
// ---------------------------------------------------------------------
__global__ __launch_bounds__(256)
void edge2_kernel(const _Float16* __restrict__ h, const int* __restrict__ ei,
                  const float* __restrict__ ea,
                  const float* __restrict__ eW, const float* __restrict__ eb,
                  float* __restrict__ aggr) {
    const long long t = (long long)blockIdx.x * 256 + threadIdx.x;
    const long long e = t >> 6;
    const int c = (int)(t & 63);
    if (e >= N_EDGES_C) return;
    const int src = ei[e];
    const int dst = ei[N_EDGES_C + e];
    const float4 a = *(const float4*)(ea + e * EDGE_DIM_C);
    const float4 w = *(const float4*)(eW + c * EDGE_DIM_C);
    float v = eb[c] + a.x * w.x + a.y * w.y + a.z * w.z + a.w * w.w;
    v += (float)h[(size_t)src * HID_C + c];
    v = fmaxf(v, 0.0f);
    atom_add_f32(&aggr[(size_t)dst * HID_C + c], v);
}

// ---------------------------------------------------------------------
// global mean pool (scatter phase): psum[g] += h2[n], cnt[g] += 1
// ---------------------------------------------------------------------
__global__ __launch_bounds__(256)
void pool_kernel(const float* __restrict__ h, const int* __restrict__ batch,
                 float* __restrict__ psum, float* __restrict__ cnt) {
    const long long t = (long long)blockIdx.x * 256 + threadIdx.x;
    const long long n = t >> 6;
    const int c = (int)(t & 63);
    if (n >= N_NODES_C) return;
    const int g = batch[n];
    atom_add_f32(&psum[(size_t)g * HID_C + c], h[(size_t)n * HID_C + c]);
    if (c == 0) atom_add_f32(&cnt[g], 1.0f);
}

// ---------------------------------------------------------------------
// final FC over pooled means: out[g,k] = mean(g) . fcW[k] + fcb[k]
// ---------------------------------------------------------------------
__global__ __launch_bounds__(256)
void fc_kernel(const float* __restrict__ psum, const float* __restrict__ cnt,
               const float* __restrict__ fcW, const float* __restrict__ fcb,
               float* __restrict__ out) {
    const int t = blockIdx.x * 256 + threadIdx.x;
    if (t >= N_GRAPHS_C * NUM_TASKS_C) return;
    const int g = t / NUM_TASKS_C;
    const int k = t - g * NUM_TASKS_C;
    const float inv = 1.0f / fmaxf(cnt[g], 1.0f);
    float s = fcb[k];
#pragma unroll
    for (int c = 0; c < HID_C; ++c)
        s += psum[(size_t)g * HID_C + c] * inv * fcW[k * HID_C + c];
    out[t] = s;
}

// ---------------------------------------------------------------------
extern "C" void kernel_launch(void* const* d_in, const int* in_sizes, int n_in,
                              void* d_out, int out_size, void* d_ws, size_t ws_size,
                              hipStream_t stream) {
    const float* x    = (const float*)d_in[0];
    const int*   ei   = (const int*)  d_in[1];   // [2, E] edge_index
    const float* ea   = (const float*)d_in[2];   // [E, 4]
    const int*   bat  = (const int*)  d_in[3];   // [N]
    const float* e1W  = (const float*)d_in[4];  const float* e1b  = (const float*)d_in[5];
    const float* n1W1 = (const float*)d_in[6];  const float* n1b1 = (const float*)d_in[7];
    const float* n1W2 = (const float*)d_in[8];  const float* n1b2 = (const float*)d_in[9];
    const float* e2W  = (const float*)d_in[10]; const float* e2b  = (const float*)d_in[11];
    const float* n2W1 = (const float*)d_in[12]; const float* n2b1 = (const float*)d_in[13];
    const float* n2W2 = (const float*)d_in[14]; const float* n2b2 = (const float*)d_in[15];
    const float* fcW  = (const float*)d_in[16]; const float* fcb  = (const float*)d_in[17];
    float* out = (float*)d_out;

    // workspace layout
    float*    aggr1 = (float*)d_ws;                              // N * 7   f32
    _Float16* h     = (_Float16*)(aggr1 + (size_t)N_NODES_C * IN_DIM_C); // N*64 f16
    float*    aggr2 = (float*)(h + (size_t)N_NODES_C * HID_C);   // N * 64  f32
    float*    h2    = aggr2 + (size_t)N_NODES_C * HID_C;         // N * 64  f32
    float*    psum  = h2    + (size_t)N_NODES_C * HID_C;         // G * 64  f32
    float*    cnt   = psum  + (size_t)N_GRAPHS_C * HID_C;        // G       f32

    hipMemsetAsync(aggr1, 0, sizeof(float) * (size_t)N_NODES_C * IN_DIM_C, stream);
    hipMemsetAsync(aggr2, 0, sizeof(float) * (size_t)N_NODES_C * HID_C,   stream);
    hipMemsetAsync(psum,  0, sizeof(float) * (size_t)N_GRAPHS_C * HID_C,  stream);
    hipMemsetAsync(cnt,   0, sizeof(float) * (size_t)N_GRAPHS_C,          stream);

    // conv1 edge phase: 8 threads / edge
    {
        const unsigned grid = (unsigned)(((long long)N_EDGES_C * 8 + 255) / 256);
        edge1_kernel<<<grid, 256, 0, stream>>>(x, ei, ea, e1W, e1b, aggr1);
    }
    // conv1 node MLP (WMMA): h(f16) = relu(mlp(x + aggr1))
    {
        const unsigned grid = (N_NODES_C + 15) / 16;
        node_mlp_kernel<IN_DIM_C, float, _Float16><<<grid, 128, 0, stream>>>(
            x, aggr1, n1W1, n1b1, n1W2, n1b2, h, N_NODES_C);
    }
    // conv2 edge phase: 64 threads / edge, f16 gather + f32 atomic scatter
    {
        const unsigned grid = (unsigned)(((long long)N_EDGES_C * 64 + 255) / 256);
        edge2_kernel<<<grid, 256, 0, stream>>>(h, ei, ea, e2W, e2b, aggr2);
    }
    // conv2 node MLP (WMMA): h2(f32) = relu(mlp(h + aggr2))
    {
        const unsigned grid = (N_NODES_C + 15) / 16;
        node_mlp_kernel<HID_C, _Float16, float><<<grid, 128, 0, stream>>>(
            h, aggr2, n2W1, n2b1, n2W2, n2b2, h2, N_NODES_C);
    }
    // pooling scatter + FC
    {
        const unsigned grid = (unsigned)(((long long)N_NODES_C * 64 + 255) / 256);
        pool_kernel<<<grid, 256, 0, stream>>>(h2, bat, psum, cnt);
    }
    {
        const unsigned grid = (N_GRAPHS_C * NUM_TASKS_C + 255) / 256;
        fc_kernel<<<grid, 256, 0, stream>>>(psum, cnt, fcW, fcb, out);
    }
}
